// TensorLayer_62113817035338
// MI455X (gfx1250) — matrix operand order repaired and checked by reading, not compiled, not run on hardware
//
#include <hip/hip_runtime.h>

typedef __attribute__((ext_vector_type(16))) _Float16 v16h;
typedef __attribute__((ext_vector_type(4)))  _Float16 v4h;
typedef __attribute__((ext_vector_type(8)))  float    v8f;
typedef __attribute__((ext_vector_type(4)))  float    v4f;

// W-fragment stream offsets in d_ws (units: halves)
#define W1_OFF 0        // 4 frags * 512 halves (one per a-block)
#define W2_OFF 2048     // 8 frags (4 Mtiles x 2 ksteps)
#define W3_OFF 6144     // 8 frags
#define W4_OFF 10240    // 2 frags (1 Mtile x 2 ksteps)
#define W_TOTAL 11264

// K index (0..31) held at position hh (0..15) of a lane's A-fragment,
// contiguous order: lanes 0-15 -> K {0..15},
// lanes 16-31 -> K {16..31}, 2 K per VGPR.
__device__ __forceinline__ int kmap32(int lane, int hh) {
    return hh + ((lane >= 16) ? 16 : 0);
}

// Pre-swizzle TT cores into per-lane A-fragment streams (f16).
// Contraction-index ordering matches the LDS staging layout:
//   GEMM1 K = i;  GEMM2 K = j*4+r1;  GEMM3 K = k*4+r2;  GEMM4 K = l*4+r3.
__global__ void tt_prep_cores(const float* __restrict__ G1,
                              const float* __restrict__ G2,
                              const float* __restrict__ G3,
                              const float* __restrict__ G4,
                              _Float16* __restrict__ W) {
    for (int h = threadIdx.x; h < W_TOTAL; h += blockDim.x) {
        float val = 0.f;
        if (h < W2_OFF) {                       // W1[(a2 r1), i], K padded to 32
            int frag = h >> 9;                  // a-block
            int r = h & 511, lane = r >> 4, hh = r & 15;
            int m = lane & 15;
            int a = frag * 4 + (m >> 2), r1 = m & 3;
            int K = kmap32(lane, hh);
            if (K < 16) val = G1[(a * 16 + K) * 4 + r1];      // G1[a,i,r1]
        } else if (h < W3_OFF) {                // W2[(b r2), (j r1)]
            int q = h - W2_OFF;
            int frag = q >> 9, mt = frag >> 1, ks = frag & 1;
            int r = q & 511, lane = r >> 4, hh = r & 15;
            int m = mt * 16 + (lane & 15);
            int b = m >> 2, r2 = m & 3;
            int K = ks * 32 + kmap32(lane, hh);
            int j = K >> 2, r1 = K & 3;
            val = G2[((b * 16 + j) * 4 + r1) * 4 + r2];       // G2[b,j,r1,r2]
        } else if (h < W4_OFF) {                // W3[(c r3), (k r2)]
            int q = h - W3_OFF;
            int frag = q >> 9, mt = frag >> 1, ks = frag & 1;
            int r = q & 511, lane = r >> 4, hh = r & 15;
            int m = mt * 16 + (lane & 15);
            int c = m >> 2, r3 = m & 3;
            int K = ks * 32 + kmap32(lane, hh);
            int k = K >> 2, r2 = K & 3;
            val = G3[((c * 16 + k) * 4 + r2) * 4 + r3];       // G3[c,k,r2,r3]
        } else {                                // W4[d, (l r3)]
            int q = h - W4_OFF;
            int ks = q >> 9;
            int r = q & 511, lane = r >> 4, hh = r & 15;
            int d = lane & 15;
            int K = ks * 32 + kmap32(lane, hh);
            int l = K >> 2, r3 = K & 3;
            val = G4[(d * 16 + l) * 4 + r3];                  // G4[d,l,r3]
        }
        W[h] = (_Float16)val;
    }
}

__global__ void __launch_bounds__(256)
tt_forward(const float* __restrict__ x,
           const _Float16* __restrict__ W,
           const float* __restrict__ bias,
           float* __restrict__ out) {
    // 320KB LDS total (ISA max per workgroup); 1 WG/WGP either way at this size.
    __shared__ __align__(32) _Float16 Xlds[4096 * 16];   // [(j,k,l)][i]       128KB
    __shared__ __align__(32) _Float16 Stage1[1024 * 64]; // [(a2,k,l)][j*4+r1] 128KB
    __shared__ __align__(32) _Float16 SB2[256 * 64];     // [(b,l)][k*4+r2]     32KB
    __shared__ __align__(32) _Float16 SB3[256 * 64];     // [(b,c)][l*4+r3]     32KB

    const int tid  = threadIdx.x;
    const int lane = tid & 31;
    const int wave = tid >> 5;
    const int lm = lane & 15;   // tile column (B/D) / tile row (A)
    const int lh = lane >> 4;   // K-half (B) / M-half (D)
    const float* xs = x + (size_t)blockIdx.x * 65536;
    float* outs = out + (size_t)blockIdx.x * 65536;

    // ---- Stage x sample into LDS once, f32->f16, transposed to [col][i].
    for (int blk = 0; blk < 64; ++blk) {
        int e = blk * 1024 + tid * 4;                 // 4 consecutive elements
        if (e + 8192 < 65536) __builtin_prefetch(xs + e + 8192, 0, 0);
        v4f xv = *(const v4f*)(xs + e);
        int i = e >> 12, n0 = e & 4095;
        #pragma unroll
        for (int q = 0; q < 4; ++q)
            Xlds[(n0 + q) * 16 + i] = (_Float16)xv[q];
    }
    __syncthreads();

    for (int ab = 0; ab < 4; ++ab) {            // a-block: a = ab*4 + a2
        // ---- GEMM1: T1[(a2 r1),(j k l)] = W1[ab](16x16,pad K32) * X(16x4096)
        {
            v16h afrag = *(const v16h*)(W + W1_OFF + ab * 512 + lane * 16);
            for (int t = wave; t < 256; t += 8) {
                int n = t * 16 + lm;            // n = j*256 + k*16 + l
                v16h bfrag = {};
                if (lh == 0)                    // lanes 16-31 hold K=16..31 -> zero pad
                    bfrag = *(const v16h*)&Xlds[n * 16];
                v8f c = {};
                c = __builtin_amdgcn_wmma_f32_16x16x32_f16(false, afrag, false, bfrag,
                                                           (short)0, c, false, false);
                int j = n >> 8, k = (n >> 4) & 15, l = n & 15;
                #pragma unroll
                for (int g = 0; g < 2; ++g) {   // D: M = v + 8*lh; pack 4 rows (r1 0..3)
                    int a2 = 2 * lh + g;
                    v4h p;
                    #pragma unroll
                    for (int q = 0; q < 4; ++q) p[q] = (_Float16)c[g * 4 + q];
                    *(v4h*)&Stage1[(a2 * 256 + k * 16 + l) * 64 + j * 4] = p;
                }
            }
        }
        __syncthreads();

        for (int a2 = 0; a2 < 4; ++a2) {
            // ---- GEMM2: T2[(b r2),(k l)] = W2(64x64) * Stage1[a2-slice](64x256)
            for (int mt = 0; mt < 4; ++mt) {
                v16h a0 = *(const v16h*)(W + W2_OFF + (mt * 2 + 0) * 512 + lane * 16);
                v16h a1 = *(const v16h*)(W + W2_OFF + (mt * 2 + 1) * 512 + lane * 16);
                for (int ct = wave; ct < 16; ct += 8) {
                    const _Float16* bp = &Stage1[(a2 * 256 + ct * 16 + lm) * 64 + lh * 16];
                    v16h b0 = *(const v16h*)(bp);
                    v16h b1 = *(const v16h*)(bp + 32);
                    v8f c = {};
                    c = __builtin_amdgcn_wmma_f32_16x16x32_f16(false, a0, false, b0,
                                                               (short)0, c, false, false);
                    c = __builtin_amdgcn_wmma_f32_16x16x32_f16(false, a1, false, b1,
                                                               (short)0, c, false, false);
                    int cc = ct * 16 + lm;      // (k,l)
                    int k = cc >> 4, l = cc & 15;
                    #pragma unroll
                    for (int g = 0; g < 2; ++g) {
                        int b = mt * 4 + 2 * lh + g;   // rows: r2 = 0..3 consecutive
                        v4h p;
                        #pragma unroll
                        for (int q = 0; q < 4; ++q) p[q] = (_Float16)c[g * 4 + q];
                        *(v4h*)&SB2[(b * 16 + l) * 64 + k * 4] = p;
                    }
                }
            }
            __syncthreads();
            // ---- GEMM3: T3[(c r3),(b l)] = W3(64x64) * SB2(64x256)
            for (int mt = 0; mt < 4; ++mt) {
                v16h a0 = *(const v16h*)(W + W3_OFF + (mt * 2 + 0) * 512 + lane * 16);
                v16h a1 = *(const v16h*)(W + W3_OFF + (mt * 2 + 1) * 512 + lane * 16);
                for (int ct = wave; ct < 16; ct += 8) {
                    const _Float16* bp = &SB2[(ct * 16 + lm) * 64 + lh * 16];
                    v16h b0 = *(const v16h*)(bp);
                    v16h b1 = *(const v16h*)(bp + 32);
                    v8f c = {};
                    c = __builtin_amdgcn_wmma_f32_16x16x32_f16(false, a0, false, b0,
                                                               (short)0, c, false, false);
                    c = __builtin_amdgcn_wmma_f32_16x16x32_f16(false, a1, false, b1,
                                                               (short)0, c, false, false);
                    int cc = ct * 16 + lm;      // (b,l)
                    int b = cc >> 4, l = cc & 15;
                    #pragma unroll
                    for (int g = 0; g < 2; ++g) {
                        int ci = mt * 4 + 2 * lh + g;  // rows: r3 = 0..3 consecutive
                        v4h p;
                        #pragma unroll
                        for (int q = 0; q < 4; ++q) p[q] = (_Float16)c[g * 4 + q];
                        *(v4h*)&SB3[(b * 16 + ci) * 64 + l * 4] = p;
                    }
                }
            }
            __syncthreads();
            // ---- GEMM4: out[d,(b c)] = W4(16x64) * SB3(64x256); +bias, ReLU, store
            {
                v16h a0 = *(const v16h*)(W + W4_OFF + 0 * 512 + lane * 16);
                v16h a1 = *(const v16h*)(W + W4_OFF + 1 * 512 + lane * 16);
                for (int ct = wave; ct < 16; ct += 8) {
                    const _Float16* bp = &SB3[(ct * 16 + lm) * 64 + lh * 16];
                    v16h b0 = *(const v16h*)(bp);
                    v16h b1 = *(const v16h*)(bp + 32);
                    v8f c = {};
                    c = __builtin_amdgcn_wmma_f32_16x16x32_f16(false, a0, false, b0,
                                                               (short)0, c, false, false);
                    c = __builtin_amdgcn_wmma_f32_16x16x32_f16(false, a1, false, b1,
                                                               (short)0, c, false, false);
                    int cc = ct * 16 + lm;      // (b,c)
                    int b = cc >> 4, ci = cc & 15;
                    int d0 = 8 * lh;            // lane holds d = d0..d0+7 contiguously
                    int ofs = (ab * 4 + a2) * 4096 + b * 256 + ci * 16 + d0;
                    v4f o0, o1;
                    #pragma unroll
                    for (int v = 0; v < 4; ++v) {
                        o0[v] = fmaxf(c[v]     + bias[ofs + v],     0.f);
                        o1[v] = fmaxf(c[v + 4] + bias[ofs + 4 + v], 0.f);
                    }
                    *(v4f*)(outs + ofs)     = o0;
                    *(v4f*)(outs + ofs + 4) = o1;
                }
            }
            __syncthreads();
        }
    }
}

extern "C" void kernel_launch(void* const* d_in, const int* in_sizes, int n_in,
                              void* d_out, int out_size, void* d_ws, size_t ws_size,
                              hipStream_t stream) {
    const float* x    = (const float*)d_in[0];
    const float* G1   = (const float*)d_in[1];
    const float* G2   = (const float*)d_in[2];
    const float* G3   = (const float*)d_in[3];
    const float* G4   = (const float*)d_in[4];
    const float* bias = (const float*)d_in[5];
    _Float16* W = (_Float16*)d_ws;
    int nsamp = in_sizes[0] / 65536;

    tt_prep_cores<<<1, 256, 0, stream>>>(G1, G2, G3, G4, W);
    tt_forward<<<nsamp, 256, 0, stream>>>(x, W, bias, (float*)d_out);
    (void)n_in; (void)out_size; (void)ws_size;
}
